// GCNModel_68186900792261
// MI455X (gfx1250) — matrix operand order repaired and checked by reading, not compiled, not run on hardware
//
#include <hip/hip_runtime.h>
#include <hip/hip_bf16.h>
#include <math.h>

typedef __attribute__((ext_vector_type(16))) _Float16 v16h;
typedef __attribute__((ext_vector_type(8)))  float    v8f;

#define F_IN   128
#define HID    128
#define NCLS   40
#define NCLS_P 48   // padded to 3 WMMA N-tiles

// Non-returning f32 atomic add: force the native L2 atomic (no CAS loop),
// device scope so results are visible to subsequent kernels in the stream.
static __device__ __forceinline__ void atomAddF(float* p, float v) {
#if defined(__gfx1250__)
    asm volatile("global_atomic_add_f32 %0, %1, off scope:SCOPE_DEV"
                 :: "v"(p), "v"(v)
                 : "memory");
#else
    __hip_atomic_fetch_add(p, v, __ATOMIC_RELAXED, __HIP_MEMORY_SCOPE_AGENT);
#endif
}

// ---------------- weight prep: transpose + f32->f16 ----------------
__global__ void prep_weights(const float* __restrict__ W1, const float* __restrict__ W2,
                             const float* __restrict__ W3,
                             _Float16* __restrict__ wt1, _Float16* __restrict__ wt2,
                             _Float16* __restrict__ wt3) {
    int idx = blockIdx.x * blockDim.x + threadIdx.x;
    if (idx < 128 * 128) {
        int n = idx >> 7, k = idx & 127;
        wt1[idx] = (_Float16)W1[k * 128 + n];   // wt[n][k] = W[k][n]
        wt2[idx] = (_Float16)W2[k * 128 + n];
    }
    if (idx < NCLS_P * 128) {
        int n = idx >> 7, k = idx & 127;
        wt3[idx] = (n < NCLS) ? (_Float16)W3[k * NCLS + n] : (_Float16)0.0f;
    }
}

// ---------------- degree / normalization ----------------
__global__ void deg_init(float* deg, int N) {
    int i = blockIdx.x * blockDim.x + threadIdx.x;
    if (i < N) deg[i] = 1.0f;   // self-loop
}
__global__ void deg_accum(const int* __restrict__ dst, float* deg, int E) {
    int e = blockIdx.x * blockDim.x + threadIdx.x;
    if (e < E) atomAddF(&deg[dst[e]], 1.0f);
}
__global__ void deg_rsqrt(float* deg, int N) {
    int i = blockIdx.x * blockDim.x + threadIdx.x;
    if (i < N) deg[i] = rsqrtf(deg[i]);   // deg >= 1 always (self-loop)
}

// ---------------- WMMA GEMM: D[M x ncols] = A[M x 128] * Wt^T ----------------
// Wt is [NTILES*16][128] f16, row n = column n of W. One wave per 16-row strip,
// A fragment reused across all NTILES wmma ops per K-step.
template <int NTILES>
__global__ void gemm_wmma(const float* __restrict__ A, const _Float16* __restrict__ Wt,
                          float* __restrict__ D, int M, int ncols, int ldo) {
    int strip = blockIdx.x * blockDim.y + threadIdx.y;
    if (strip * 16 >= M) return;
    int lane = threadIdx.x;
    int r = lane & 15;        // row within tile (A) / col within tile (B,D)
    int hi = lane >> 4;       // K-half selector

    const float* arow = A + (size_t)(strip * 16 + r) * F_IN;

    v8f zero = {0.f, 0.f, 0.f, 0.f, 0.f, 0.f, 0.f, 0.f};
    v8f acc[NTILES];
#pragma unroll
    for (int nt = 0; nt < NTILES; ++nt) acc[nt] = zero;

#pragma unroll
    for (int ks = 0; ks < F_IN / 32; ++ks) {
        int base = ks * 32;
        // A 16x32 f16 fragment: halves 0..15 = K base+16*hi.. (contiguous)
        const float4* p0 = (const float4*)(arow + base + 16 * hi);
        const float4* p1 = (const float4*)(arow + base + 16 * hi + 8);
        float4 x0 = p0[0], x1 = p0[1];
        float4 x2 = p1[0], x3 = p1[1];
        v16h a;
        a[0]  = (_Float16)x0.x; a[1]  = (_Float16)x0.y; a[2]  = (_Float16)x0.z; a[3]  = (_Float16)x0.w;
        a[4]  = (_Float16)x1.x; a[5]  = (_Float16)x1.y; a[6]  = (_Float16)x1.z; a[7]  = (_Float16)x1.w;
        a[8]  = (_Float16)x2.x; a[9]  = (_Float16)x2.y; a[10] = (_Float16)x2.z; a[11] = (_Float16)x2.w;
        a[12] = (_Float16)x3.x; a[13] = (_Float16)x3.y; a[14] = (_Float16)x3.z; a[15] = (_Float16)x3.w;
#pragma unroll
        for (int nt = 0; nt < NTILES; ++nt) {
            // B 32x16 fragment: half i = B[K = base + 16*hi + i][n = nt*16+r]
            //                   = Wt[nt*16+r][base + 16*hi + i] (contiguous)
            v16h b = *(const v16h*)(Wt + (size_t)(nt * 16 + r) * F_IN + base + 16 * hi);
            acc[nt] = __builtin_amdgcn_wmma_f32_16x16x32_f16(
                false, a, false, b, (short)0, acc[nt], false, false);
        }
    }
    // D layout: VGPR j -> row j + 8*hi, col = lane&15
#pragma unroll
    for (int nt = 0; nt < NTILES; ++nt) {
        int cc = nt * 16 + r;
        if (cc < ncols) {
            float* dcol = D + (size_t)(strip * 16 + 8 * hi) * ldo + cc;
#pragma unroll
            for (int j = 0; j < 8; ++j) dcol[(size_t)j * ldo] = acc[nt][j];
        }
    }
}

// ---------------- aggregation: self-loop init then edge scatter ----------------
template <int F>
__global__ void agg_init(const float* __restrict__ xw, const float* __restrict__ dinv,
                         float* __restrict__ agg, int N) {
    size_t i = (size_t)blockIdx.x * blockDim.x + threadIdx.x;
    size_t tot = (size_t)N * F;
    if (i < tot) {
        int n = (int)(i / F);
        float w = dinv[n];
        agg[i] = w * w * xw[i];   // self-loop contribution: dinv[n]*1*dinv[n]*xw[n]
    }
}

// one wave per edge, 128 features = one float4 per lane
__global__ void scatter128(const int* __restrict__ src, const int* __restrict__ dst,
                           const float* __restrict__ dinv, const float* __restrict__ xw,
                           float* __restrict__ agg, int E) {
    int e = blockIdx.x * blockDim.y + threadIdx.y;
    if (e >= E) return;
    int s = src[e], d = dst[e];
    float w = dinv[s] * dinv[d];
    int lane = threadIdx.x;
    float4 v = ((const float4*)(xw + (size_t)s * 128))[lane];
    float* p = agg + (size_t)d * 128 + lane * 4;
    atomAddF(p + 0, v.x * w);
    atomAddF(p + 1, v.y * w);
    atomAddF(p + 2, v.z * w);
    atomAddF(p + 3, v.w * w);
}

// one wave per edge, 40 features: lane handles f=lane and (lane<8) f=32+lane
__global__ void scatter40(const int* __restrict__ src, const int* __restrict__ dst,
                          const float* __restrict__ dinv, const float* __restrict__ xw,
                          float* __restrict__ agg, int E) {
    int e = blockIdx.x * blockDim.y + threadIdx.y;
    if (e >= E) return;
    int s = src[e], d = dst[e];
    float w = dinv[s] * dinv[d];
    int lane = threadIdx.x;
    const float* xr = xw + (size_t)s * NCLS;
    float* p = agg + (size_t)d * NCLS;
    atomAddF(p + lane, xr[lane] * w);
    if (lane < 8) atomAddF(p + 32 + lane, xr[32 + lane] * w);
}

__global__ void bias_relu(float* __restrict__ h, const float* __restrict__ b, int N) {
    size_t i = (size_t)blockIdx.x * blockDim.x + threadIdx.x;
    if (i < (size_t)N * HID) {
        float v = h[i] + b[i & (HID - 1)];
        h[i] = v > 0.0f ? v : 0.0f;
    }
}

// ---------------- bias + log-softmax over 40 classes, one wave per node ----------------
__global__ void log_softmax40(const float* __restrict__ agg, const float* __restrict__ b,
                              float* __restrict__ out, int N) {
    int n = blockIdx.x * blockDim.y + threadIdx.y;
    if (n >= N) return;
    int lane = threadIdx.x;
    const float* row = agg + (size_t)n * NCLS;
    float v0 = row[lane] + b[lane];                                    // f = lane (always < 40)
    float v1 = (lane < 8) ? (row[32 + lane] + b[32 + lane]) : -INFINITY;
    float m = fmaxf(v0, v1);
#pragma unroll
    for (int off = 16; off; off >>= 1) m = fmaxf(m, __shfl_xor(m, off, 32));
    float s = expf(v0 - m) + ((lane < 8) ? expf(v1 - m) : 0.0f);
#pragma unroll
    for (int off = 16; off; off >>= 1) s += __shfl_xor(s, off, 32);
    float ls = m + logf(s);
    float* orow = out + (size_t)n * NCLS;
    orow[lane] = v0 - ls;
    if (lane < 8) orow[32 + lane] = v1 - ls;
}

extern "C" void kernel_launch(void* const* d_in, const int* in_sizes, int n_in,
                              void* d_out, int out_size, void* d_ws, size_t ws_size,
                              hipStream_t stream) {
    const float* x  = (const float*)d_in[0];
    const float* W1 = (const float*)d_in[1];
    const float* b1 = (const float*)d_in[2];
    const float* W2 = (const float*)d_in[3];
    const float* b2 = (const float*)d_in[4];
    const float* W3 = (const float*)d_in[5];
    const float* b3 = (const float*)d_in[6];
    const int*   ei = (const int*)d_in[7];
    int N = in_sizes[0] / F_IN;
    int E = in_sizes[7] / 2;
    const int* srcv = ei;
    const int* dstv = ei + E;
    float* out = (float*)d_out;

    // workspace carve-out (256B aligned)
    char* ws = (char*)d_ws;
    size_t off = 0;
    auto carve = [&](size_t bytes) -> void* {
        void* p = ws + off;
        off += (bytes + 255) & ~(size_t)255;
        return p;
    };
    float*    dinv = (float*)carve((size_t)N * 4);
    float*    bufA = (float*)carve((size_t)N * HID * 4);
    float*    bufB = (float*)carve((size_t)N * HID * 4);
    _Float16* wt1  = (_Float16*)carve(128 * 128 * 2);
    _Float16* wt2  = (_Float16*)carve(128 * 128 * 2);
    _Float16* wt3  = (_Float16*)carve(NCLS_P * 128 * 2);
    float* xw3  = bufA;                       // N*40 floats
    float* agg3 = bufA + (size_t)N * 64;      // disjoint second half of bufA

    dim3 w32x8(32, 8);
    int strips     = (N + 15) / 16;
    int gemmBlocks = (strips + 7) / 8;
    int edgeBlocks = (E + 7) / 8;
    size_t totF    = (size_t)N * HID;
    int elemBlocks = (int)((totF + 255) / 256);
    size_t tot40   = (size_t)N * NCLS;
    int elem40     = (int)((tot40 + 255) / 256);

    // norm + weights prep
    prep_weights<<<(128 * 128 + 255) / 256, 256, 0, stream>>>(W1, W2, W3, wt1, wt2, wt3);
    deg_init<<<(N + 255) / 256, 256, 0, stream>>>(dinv, N);
    deg_accum<<<(E + 255) / 256, 256, 0, stream>>>(dstv, dinv, E);
    deg_rsqrt<<<(N + 255) / 256, 256, 0, stream>>>(dinv, N);

    // layer 1
    gemm_wmma<8><<<gemmBlocks, w32x8, 0, stream>>>(x, wt1, bufA, N, 128, 128);
    agg_init<128><<<elemBlocks, 256, 0, stream>>>(bufA, dinv, bufB, N);
    scatter128<<<edgeBlocks, w32x8, 0, stream>>>(srcv, dstv, dinv, bufA, bufB, E);
    bias_relu<<<elemBlocks, 256, 0, stream>>>(bufB, b1, N);

    // layer 2
    gemm_wmma<8><<<gemmBlocks, w32x8, 0, stream>>>(bufB, wt2, bufA, N, 128, 128);
    agg_init<128><<<elemBlocks, 256, 0, stream>>>(bufA, dinv, bufB, N);
    scatter128<<<edgeBlocks, w32x8, 0, stream>>>(srcv, dstv, dinv, bufA, bufB, E);
    bias_relu<<<elemBlocks, 256, 0, stream>>>(bufB, b2, N);

    // layer 3 + log-softmax
    gemm_wmma<3><<<gemmBlocks, w32x8, 0, stream>>>(bufB, wt3, xw3, N, NCLS, NCLS);
    agg_init<40><<<elem40, 256, 0, stream>>>(xw3, dinv, agg3, N);
    scatter40<<<edgeBlocks, w32x8, 0, stream>>>(srcv, dstv, dinv, xw3, agg3, E);
    log_softmax40<<<(N + 7) / 8, w32x8, 0, stream>>>(agg3, b3, out, N);
}
